// TextGenerator_38371237823113
// MI455X (gfx1250) — matrix operand order repaired and checked by reading, not compiled, not run on hardware
//
#include <hip/hip_runtime.h>
#include <hip/hip_bf16.h>
#include <math.h>
#include <stdint.h>

#define DEV __device__ __forceinline__

typedef __bf16 bf16_t;
typedef __attribute__((ext_vector_type(16))) __bf16 v16bf;
typedef __attribute__((ext_vector_type(8)))  __bf16 v8bf;
typedef __attribute__((ext_vector_type(4)))  __bf16 v4bf;
typedef __attribute__((ext_vector_type(8)))  float  v8f;

static constexpr int V  = 50257;
static constexpr int H  = 600;
static constexpr int B  = 128;
static constexpr int R  = 200;
static constexpr int H2 = 1200;
static constexpr int H4 = 2400;
static constexpr int NT_V   = 3142;          // ceil(V/16)
static constexpr int NPAD_V = NT_V * 16;     // 50272
static constexpr int K_NH   = 1216;          // 1200 padded to mult of 32
static constexpr int KPAD_H = 608;           // 600 padded to mult of 32

// ---------------------------------------------------------------- helpers ---
DEV bf16_t f2bf(float f) { return (bf16_t)f; }   // RNE fptrunc -> v_cvt bf16
DEV float sigmoid_f(float x) { return 1.f / (1.f + __expf(-x)); }

// Async global->LDS copy (ISA 15.18.3 GLOBAL_LOAD_ASYNC_TO_LDS_B128, ASYNCcnt).
// lds_off is the wave-relative LDS byte offset = low 32 bits of the generic
// pointer to a __shared__ object (flat-LDS aperture truncation, ISA 10.2).
DEV void async_g2l_b128(unsigned lds_off, const void* gaddr) {
  asm volatile("global_load_async_to_lds_b128 %0, %1, off"
               :: "v"(lds_off), "v"(gaddr) : "memory");
}
DEV void wait_async0() {
#if __has_builtin(__builtin_amdgcn_s_wait_asynccnt)
  __builtin_amdgcn_s_wait_asynccnt(0);
#else
  asm volatile("s_wait_asynccnt 0x0" ::: "memory");
#endif
}

// A operand (16x32 slab, ISA 7.12.2): lane m = lane&15, h16 = lane>>4.
// element e : k = k0 + h16*16 + e (contiguous, matches B operand).
// Unconditional (caller guarantees padded K).
DEV v16bf a_frag_bf16(const bf16_t* row, int k0, int h16) {
  int b1 = k0 + h16 * 16;
  v8bf lo = *(const v8bf*)(row + b1);
  v8bf hi = *(const v8bf*)(row + b1 + 8);
  v16bf f;
#pragma unroll
  for (int e = 0; e < 8; ++e) { f[e] = lo[e]; f[8 + e] = hi[e]; }
  return f;
}

// B operand (32x16 slab): B[k][n] = W[n][k]; lane n reads 16 contiguous k at
// k = k0 + (lane>>4)*16 + e.
DEV v16bf b_frag_bf16(const bf16_t* row, int k0, int kh) {
  int b0 = k0 + kh * 16;
  v8bf lo = *(const v8bf*)(row + b0);
  v8bf hi = *(const v8bf*)(row + b0 + 8);
  v16bf f;
#pragma unroll
  for (int e = 0; e < 8; ++e) { f[e] = lo[e]; f[8 + e] = hi[e]; }
  return f;
}

DEV v16bf b_frag_f32(const float* row, int k0, int kh) {   // full tiles only
  const float4* p = (const float4*)(row + k0 + kh * 16);
  float4 x0 = p[0], x1 = p[1], x2 = p[2], x3 = p[3];
  v16bf f;
  f[0] =f2bf(x0.x); f[1] =f2bf(x0.y); f[2] =f2bf(x0.z); f[3] =f2bf(x0.w);
  f[4] =f2bf(x1.x); f[5] =f2bf(x1.y); f[6] =f2bf(x1.z); f[7] =f2bf(x1.w);
  f[8] =f2bf(x2.x); f[9] =f2bf(x2.y); f[10]=f2bf(x2.z); f[11]=f2bf(x2.w);
  f[12]=f2bf(x3.x); f[13]=f2bf(x3.y); f[14]=f2bf(x3.z); f[15]=f2bf(x3.w);
  return f;
}

DEV v8f wmma_bf16(v16bf a, v16bf b, v8f c) {
  return __builtin_amdgcn_wmma_f32_16x16x32_bf16(false, a, false, b, (short)0, c, false, false);
}

// ------------------------------------------------------------- kernel 0 -----
// Pack bf16, zero-padded operands for the LSTM GEMM:
//   wih_bf[dir][n][0:608]  <- W_ih[dir][n][0:600]
//   x_bf[m][0:608]         <- emb[word[m]][0:600]
__global__ void prep_lstm(const int* __restrict__ word, const float* __restrict__ emb,
                          const float* __restrict__ Wf, const float* __restrict__ Wb,
                          bf16_t* __restrict__ wih_bf, bf16_t* __restrict__ x_bf) {
  int rid = blockIdx.x;
  const float* src;
  bf16_t* dst;
  if (rid < 2 * H4) {
    int dir = rid / H4, n = rid - dir * H4;
    src = (dir ? Wb : Wf) + (size_t)n * H;
    dst = wih_bf + (size_t)rid * KPAD_H;
  } else {
    int m = rid - 2 * H4;
    src = emb + (size_t)word[m] * H;
    dst = x_bf + (size_t)m * KPAD_H;
  }
  for (int k = threadIdx.x; k < KPAD_H; k += 64)
    dst[k] = f2bf(k < H ? src[k] : 0.f);
}

// ------------------------------------------------------------- kernel 1 -----
// g[dir][m][n] = x . W_ih[dir][n] + b_ih[n] + b_hh[n]
// grid (150 n-tiles, 2 dirs), block 256 = 8 waves = 8 m-tiles (M=128).
__global__ void lstm_gemm(const bf16_t* __restrict__ x_bf, const bf16_t* __restrict__ wih_bf,
                          const float* __restrict__ bihf, const float* __restrict__ bhhf,
                          const float* __restrict__ bihb, const float* __restrict__ bhhb,
                          float* __restrict__ g) {
  int lane = threadIdx.x & 31, wave = threadIdx.x >> 5, h16 = lane >> 4;
  int nt = blockIdx.x, dir = blockIdx.y;
  int m = wave * 16 + (lane & 15);
  int n = nt * 16 + (lane & 15);
  const bf16_t* arow = x_bf + (size_t)m * KPAD_H;
  const bf16_t* brow = wih_bf + ((size_t)dir * H4 + n) * KPAD_H;
  v8f acc = {};
#pragma unroll 2
  for (int k0 = 0; k0 < KPAD_H; k0 += 32)
    acc = wmma_bf16(a_frag_bf16(arow, k0, h16), b_frag_bf16(brow, k0, h16), acc);
  const float* bi = dir ? bihb : bihf;
  const float* bh = dir ? bhhb : bhhf;
  float bias = bi[n] + bh[n];
  float* gout = g + (size_t)dir * B * H4;
#pragma unroll
  for (int i = 0; i < 8; ++i) {
    int mm = wave * 16 + i + 8 * h16;
    gout[(size_t)mm * H4 + n] = acc[i] + bias;
  }
}

// ------------------------------------------------------------- kernel 2 -----
// gates -> h, new_cell, bf16 cat[:,0:1200], and hbl[b] = h[b].b_lin
__global__ void lstm_pointwise(const float* __restrict__ g, const float* __restrict__ b_lin,
                               float* __restrict__ h, float* __restrict__ hbl,
                               bf16_t* __restrict__ cat_bf, float* __restrict__ new_cell) {
  int b = blockIdx.x, t = threadIdx.x;
  __shared__ float red[256];
  float part = 0.f;
  for (int n = t; n < H; n += 256) {
#pragma unroll
    for (int dir = 0; dir < 2; ++dir) {
      const float* gr = g + ((size_t)dir * B + b) * H4;
      float gi = gr[n], gg = gr[2 * H + n], go = gr[3 * H + n];
      float c  = sigmoid_f(gi) * tanhf(gg);
      float hh = sigmoid_f(go) * tanhf(c);
      new_cell[((size_t)dir * B + b) * H + n] = c;
      int d = dir * H + n;
      h[(size_t)b * H2 + d]      = hh;
      cat_bf[(size_t)b * H4 + d] = f2bf(hh);
      part += hh * b_lin[d];
    }
  }
  red[t] = part;
  __syncthreads();
  for (int off = 128; off > 0; off >>= 1) {
    if (t < off) red[t] += red[t + off];
    __syncthreads();
  }
  if (t == 0) hbl[b] = red[0];
}

// ------------------------------------------------------------- kernel 3 -----
// hW[b][k] = sum_d h[b][d] * W_lin[d][k]   (tiny; memory-trivial)
__global__ void hw_gemv(const float* __restrict__ h, const float* __restrict__ W_lin,
                        float* __restrict__ hW) {
  int bb = blockIdx.y;
  int k = blockIdx.x * 256 + threadIdx.x;
  __shared__ float sh[H2];
  for (int d = threadIdx.x; d < H2; d += 256) sh[d] = h[(size_t)bb * H2 + d];
  __syncthreads();
  if (k < H2) {
    float s = 0.f;
    for (int d = 0; d < H2; ++d) s = fmaf(sh[d], W_lin[(size_t)d * H2 + k], s);
    hW[(size_t)bb * H2 + k] = s;
  }
}

// ------------------------------------------------------------- kernel 4 -----
// per-b: scores = hW[b].encoded[b,r] + hbl[b]; log_softmax over R;
// selected[d] = sum_r att[r]*encoded[b,r,d] -> bf16 cat[:,1200:2400]
// hW row staged to LDS with GLOBAL_LOAD_ASYNC_TO_LDS_B128 (ASYNCcnt path).
__global__ void attention_kernel(const float* __restrict__ encoded, const float* __restrict__ hW,
                                 const float* __restrict__ hbl, float* __restrict__ log_att,
                                 bf16_t* __restrict__ cat_bf) {
  int b = blockIdx.x, t = threadIdx.x, lane = t & 31, wave = t >> 5;
  __shared__ float s_hw[H2];
  __shared__ float s_sc[256];
  __shared__ float s_red[256];
  for (int q = t; q < H2 / 4; q += 256) {
    unsigned loff = (unsigned)(uintptr_t)&s_hw[q * 4];
    async_g2l_b128(loff, hW + (size_t)b * H2 + q * 4);
  }
  wait_async0();
  __syncthreads();
  const float* enc = encoded + (size_t)b * R * H2;
  float hb = hbl[b];
  const float4* sh4 = (const float4*)s_hw;
  for (int r = wave; r < R; r += 8) {
    const float4* er4 = (const float4*)(enc + (size_t)r * H2);
    float p = 0.f;
    for (int q = lane; q < H2 / 4; q += 32) {
      float4 a = sh4[q];
      float4 e = er4[q];
      p = fmaf(a.x, e.x, fmaf(a.y, e.y, fmaf(a.z, e.z, fmaf(a.w, e.w, p))));
    }
    for (int off = 16; off > 0; off >>= 1) p += __shfl_down(p, off);
    if (lane == 0) s_sc[r] = p + hb;
  }
  __syncthreads();
  s_red[t] = (t < R) ? s_sc[t] : -INFINITY;
  __syncthreads();
  for (int off = 128; off > 0; off >>= 1) {
    if (t < off) s_red[t] = fmaxf(s_red[t], s_red[t + off]);
    __syncthreads();
  }
  float mx = s_red[0];
  __syncthreads();
  s_red[t] = (t < R) ? __expf(s_sc[t] - mx) : 0.f;
  __syncthreads();
  for (int off = 128; off > 0; off >>= 1) {
    if (t < off) s_red[t] += s_red[t + off];
    __syncthreads();
  }
  float lse = __logf(s_red[0]);
  __syncthreads();
  if (t < R) {
    float la = s_sc[t] - mx - lse;
    log_att[(size_t)b * R + t] = la;
    s_sc[t] = __expf(la);
  }
  __syncthreads();
  for (int d = t; d < H2; d += 256) {
    float acc = 0.f;
    for (int r = 0; r < R; ++r) acc = fmaf(s_sc[r], enc[(size_t)r * H2 + d], acc);
    cat_bf[(size_t)b * H4 + H2 + d] = f2bf(acc);
  }
}

// ------------------------------------------------------------- kernel 5 -----
// nh = tanh(cat @ W_tanh.T + b_tanh); writes nh (fp32), nh_bf, nh_out (2,B,H)
// grid 75 n-tiles, block 256 = 8 waves = 8 m-tiles.  K=2400, no tails.
__global__ void tanh_gemm(const bf16_t* __restrict__ cat_bf, const float* __restrict__ W_tanh,
                          const float* __restrict__ b_tanh, float* __restrict__ nh,
                          bf16_t* __restrict__ nh_bf, float* __restrict__ nh_out) {
  int lane = threadIdx.x & 31, wave = threadIdx.x >> 5, h16 = lane >> 4;
  int nt = blockIdx.x;
  int m = wave * 16 + (lane & 15);
  int n = nt * 16 + (lane & 15);
  const bf16_t* arow = cat_bf + (size_t)m * H4;
  const float*  brow = W_tanh + (size_t)n * H4;
  v8f acc = {};
#pragma unroll 2
  for (int k0 = 0; k0 < H4; k0 += 32)
    acc = wmma_bf16(a_frag_bf16(arow, k0, h16), b_frag_f32(brow, k0, h16), acc);
  float bias = b_tanh[n];
  int s = n / H, hh = n - s * H;                 // nh_out[s][b][hh] = nh[b][n]
#pragma unroll
  for (int i = 0; i < 8; ++i) {
    int mm = wave * 16 + i + 8 * h16;
    float v = tanhf(acc[i] + bias);
    nh[(size_t)mm * H2 + n]      = v;
    nh_bf[(size_t)mm * K_NH + n] = f2bf(v);
    nh_out[((size_t)s * B + mm) * H + hh] = v;
  }
}

// ------------------------------------------------------------- kernel 6 -----
// logits = nh @ W_soft.T + b_soft.  One block per 16-col vocab tile; W_soft
// tile staged fp32->bf16 into LDS once (HBM read exactly once); 8 waves cover
// M=128 from the shared tile.  nh_bf is zero-padded -> no K predication.
__global__ void soft_gemm(const bf16_t* __restrict__ nh_bf, const float* __restrict__ W_soft,
                          const float* __restrict__ b_soft, float* __restrict__ logits) {
  __shared__ bf16_t sB[16][72];                  // 144B pitch: 16B-aligned rows
  int t = threadIdx.x, lane = t & 31, wave = t >> 5, h16 = lane >> 4;
  int nt = blockIdx.x;
  int nloc = lane & 15;
  int n = nt * 16 + nloc;
  int m = wave * 16 + (lane & 15);
  const bf16_t* arow = nh_bf + (size_t)m * K_NH;

  // staging role: 16 rows x 16 float4-columns
  int rr = t >> 4, c4 = t & 15;
  int gn = nt * 16 + rr;
  const float* wrow = W_soft + (size_t)gn * H2;

  v8f acc = {};
#pragma unroll 1
  for (int kc = 0; kc < K_NH; kc += 64) {
    __syncthreads();
    int gk = kc + c4 * 4;
    v4bf outv;
    if (gn < V && gk + 4 <= H2) {
      float4 x = *(const float4*)(wrow + gk);
      outv[0] = f2bf(x.x); outv[1] = f2bf(x.y); outv[2] = f2bf(x.z); outv[3] = f2bf(x.w);
      if (gk + 64 < H2) __builtin_prefetch(wrow + gk + 64, 0, 1);
    } else {
#pragma unroll
      for (int j = 0; j < 4; ++j) {
        int k = gk + j;
        outv[j] = f2bf((gn < V && k < H2) ? wrow[k] : 0.f);
      }
    }
    *(v4bf*)&sB[rr][c4 * 4] = outv;
    __syncthreads();
#pragma unroll
    for (int ks = 0; ks < 64; ks += 32) {
      v16bf a = a_frag_bf16(arow, kc + ks, h16);
      const v8bf* bp = (const v8bf*)&sB[nloc][ks + h16 * 16];
      v8bf blo = bp[0], bhi = bp[1];
      v16bf bfr;
#pragma unroll
      for (int e = 0; e < 8; ++e) { bfr[e] = blo[e]; bfr[8 + e] = bhi[e]; }
      acc = wmma_bf16(a, bfr, acc);
    }
  }
  if (n < V) {
    float bias = b_soft[n];
#pragma unroll
    for (int i = 0; i < 8; ++i) {
      int mm = wave * 16 + i + 8 * h16;
      logits[(size_t)mm * NPAD_V + n] = acc[i] + bias;
    }
  }
}

// ------------------------------------------------------------- kernel 7 -----
__global__ void pcopy_kernel(const float* __restrict__ nh, const float* __restrict__ W_sig,
                             const float* __restrict__ b_sig, float* __restrict__ p_copy) {
  int b = blockIdx.x, t = threadIdx.x;
  __shared__ float red[128];
  float p = 0.f;
  for (int d = t; d < H2; d += 128) p = fmaf(nh[(size_t)b * H2 + d], W_sig[d], p);
  red[t] = p;
  __syncthreads();
  for (int off = 64; off > 0; off >>= 1) {
    if (t < off) red[t] += red[t + off];
    __syncthreads();
  }
  if (t == 0) p_copy[b] = sigmoid_f(red[0] + b_sig[0]);
}

// ------------------------------------------------------------- kernel 8 -----
// row-wise log_softmax over V (logits stay in L2)
__global__ void logsoftmax_kernel(const float* __restrict__ logits, float* __restrict__ out) {
  int b = blockIdx.x, t = threadIdx.x;
  __shared__ float red[256];
  const float* row = logits + (size_t)b * NPAD_V;
  float mx = -INFINITY;
  for (int v = t; v < V; v += 256) mx = fmaxf(mx, row[v]);
  red[t] = mx;
  __syncthreads();
  for (int off = 128; off > 0; off >>= 1) {
    if (t < off) red[t] = fmaxf(red[t], red[t + off]);
    __syncthreads();
  }
  mx = red[0];
  __syncthreads();
  float s = 0.f;
  for (int v = t; v < V; v += 256) s += __expf(row[v] - mx);
  red[t] = s;
  __syncthreads();
  for (int off = 128; off > 0; off >>= 1) {
    if (t < off) red[t] += red[t + off];
    __syncthreads();
  }
  float lse = mx + __logf(red[0]);
  for (int v = t; v < V; v += 256) out[(size_t)b * V + v] = row[v] - lse;
}

// -------------------------------------------------------------- launcher ----
extern "C" void kernel_launch(void* const* d_in, const int* in_sizes, int n_in,
                              void* d_out, int out_size, void* d_ws, size_t ws_size,
                              hipStream_t stream) {
  (void)in_sizes; (void)n_in; (void)out_size; (void)ws_size;

  const int*   word    = (const int*)  d_in[0];
  // d_in[1] hidden, d_in[2] cell: unused by forward (reference keeps that bug)
  const float* encoded = (const float*)d_in[3];
  const float* emb     = (const float*)d_in[4];
  const float* Wihf    = (const float*)d_in[5];
  const float* bihf    = (const float*)d_in[7];
  const float* bhhf    = (const float*)d_in[8];
  const float* Wihb    = (const float*)d_in[9];
  const float* bihb    = (const float*)d_in[11];
  const float* bhhb    = (const float*)d_in[12];
  const float* Wlin    = (const float*)d_in[13];
  const float* blin    = (const float*)d_in[14];
  const float* Wtanh   = (const float*)d_in[15];
  const float* btanh   = (const float*)d_in[16];
  const float* Wsoft   = (const float*)d_in[17];
  const float* bsoft   = (const float*)d_in[18];
  const float* Wsig    = (const float*)d_in[19];
  const float* bsig    = (const float*)d_in[20];

  // d_out: out_prob[B,V] | log_att[B,R] | p_copy[B,1] | nh_out[2,B,H] | new_cell[2,B,H]
  float* out_prob = (float*)d_out;
  float* log_att  = out_prob + (size_t)B * V;
  float* p_copy   = log_att + (size_t)B * R;
  float* nh_out   = p_copy + B;
  float* new_cell = nh_out + (size_t)2 * B * H;

  char* w = (char*)d_ws;
  auto take = [&](size_t bytes) -> void* {
    void* p = (void*)w;
    w += (bytes + 255) & ~(size_t)255;
    return p;
  };
  bf16_t* wih_bf = (bf16_t*)take(sizeof(bf16_t) * 2 * H4 * KPAD_H);
  bf16_t* x_bf   = (bf16_t*)take(sizeof(bf16_t) * B * KPAD_H);
  float*  g      = (float*) take(sizeof(float)  * 2 * B * H4);
  float*  h      = (float*) take(sizeof(float)  * B * H2);
  float*  hbl    = (float*) take(sizeof(float)  * B);
  float*  hW     = (float*) take(sizeof(float)  * B * H2);
  float*  nh     = (float*) take(sizeof(float)  * B * H2);
  bf16_t* cat_bf = (bf16_t*)take(sizeof(bf16_t) * B * H4);
  bf16_t* nh_bf  = (bf16_t*)take(sizeof(bf16_t) * B * K_NH);
  float*  logits = (float*) take(sizeof(float)  * B * NPAD_V);

  prep_lstm<<<dim3(2 * H4 + B), 64, 0, stream>>>(word, emb, Wihf, Wihb, wih_bf, x_bf);
  lstm_gemm<<<dim3(H4 / 16, 2), 256, 0, stream>>>(x_bf, wih_bf, bihf, bhhf, bihb, bhhb, g);
  lstm_pointwise<<<dim3(B), 256, 0, stream>>>(g, blin, h, hbl, cat_bf, new_cell);
  hw_gemv<<<dim3((H2 + 255) / 256, B), 256, 0, stream>>>(h, Wlin, hW);
  attention_kernel<<<dim3(B), 256, 0, stream>>>(encoded, hW, hbl, log_att, cat_bf);
  hipMemsetAsync(nh_bf, 0, sizeof(bf16_t) * B * K_NH, stream);   // zero K-pad
  tanh_gemm<<<dim3(H2 / 16), 256, 0, stream>>>(cat_bf, Wtanh, btanh, nh, nh_bf, nh_out);
  soft_gemm<<<dim3(NT_V), 256, 0, stream>>>(nh_bf, Wsoft, bsoft, logits);
  pcopy_kernel<<<dim3(B), 128, 0, stream>>>(nh, Wsig, bsig, p_copy);
  logsoftmax_kernel<<<dim3(B), 256, 0, stream>>>(logits, out_prob);
}
